// CNN_Encoder_33328946217622
// MI455X (gfx1250) — matrix operand order repaired and checked by reading, not compiled, not run on hardware
//
#include <hip/hip_runtime.h>

// ---------------------------------------------------------------------------
// CDNA5 (gfx1250) implicit-GEMM multi-scale conv encoder.
// wave32, v_wmma_f32_16x16x32_bf16, async global->LDS double buffering.
// ---------------------------------------------------------------------------

typedef __attribute__((ext_vector_type(16))) __bf16 v16bf;
typedef __attribute__((ext_vector_type(8)))  __bf16 v8bf;
typedef __attribute__((ext_vector_type(8)))  float  v8f;

#define BN_EPS 1e-5f

// Geometry
constexpr int Bn = 32;    // batch
constexpr int Lc = 144;   // input channels
constexpr int Dc = 96;    // output channels
constexpr int Hs = 64;    // height
constexpr int Ws = 64;    // width
constexpr int Lp = 160;   // channels padded to multiple of 32
constexpr int Hp = 70;    // padded height  (+3 halo each side)
constexpr int Wp = 72;    // padded width   (+3 halo, +2 pad to 72)
constexpr int KG_TOTAL = 9 + 25 + 49;  // 83 (kh,kw) taps across 3 branches
constexpr int CHUNK_ELEMS = 7 * 72 * 32;      // LDS slab elems per 32-ch chunk
constexpr int FILL_UNITS  = 7 * 72 * 4;       // 16-byte units per slab (2016)

// ---------------------------------------------------------------------------
// Stage 1: x (NCHW f32) -> xP (padded NHWC bf16: [b][hp 70][wp 72][c 160],
// zeros in halo rows/cols and channels 144..159)
// ---------------------------------------------------------------------------
__global__ void prep_x_kernel(const float* __restrict__ x, __bf16* __restrict__ xP) {
  long long idx = (long long)blockIdx.x * blockDim.x + threadIdx.x;
  const long long total = (long long)Bn * Hp * Wp * Lp;
  if (idx >= total) return;
  int c = (int)(idx % Lp);
  long long r = idx / Lp;
  int wp = (int)(r % Wp); r /= Wp;
  int hp = (int)(r % Hp);
  int b  = (int)(r / Hp);
  int h = hp - 3, w = wp - 3;
  float v = 0.f;
  if (c < Lc && h >= 0 && h < Hs && w >= 0 && w < Ws)
    v = x[(((long long)b * Lc + c) * Hs + h) * Ws + w];
  xP[idx] = (__bf16)v;
}

// ---------------------------------------------------------------------------
// Stage 2: fold BN scale into weights; layout Wt[kg][d][c], c padded -> 160
// kg: 0..8 = 3x3 taps, 9..33 = 5x5 taps, 34..82 = 7x7 taps
// ---------------------------------------------------------------------------
__global__ void prep_w_kernel(const float* __restrict__ w1, const float* __restrict__ g1, const float* __restrict__ v1,
                              const float* __restrict__ w2, const float* __restrict__ g2, const float* __restrict__ v2,
                              const float* __restrict__ w3, const float* __restrict__ g3, const float* __restrict__ v3,
                              __bf16* __restrict__ Wt) {
  int idx = blockIdx.x * blockDim.x + threadIdx.x;
  const int total = KG_TOTAL * Dc * Lp;
  if (idx >= total) return;
  int c = idx % Lp;
  int rest = idx / Lp;
  int d = rest % Dc;
  int kg = rest / Dc;
  float val = 0.f;
  if (c < Lc) {
    const float *w, *g, *v;
    int S, kpos;
    if (kg < 9)       { w = w1; g = g1; v = v1; S = 3; kpos = kg; }
    else if (kg < 34) { w = w2; g = g2; v = v2; S = 5; kpos = kg - 9; }
    else              { w = w3; g = g3; v = v3; S = 7; kpos = kg - 34; }
    int kh = kpos / S, kw = kpos % S;
    float inv = g[d] * rsqrtf(v[d] + BN_EPS);
    val = w[(((long long)d * Lc + c) * S + kh) * S + kw] * inv;
  }
  Wt[idx] = (__bf16)val;
}

// biasF[br*96 + d] = b*inv + beta - mean*inv
__global__ void prep_bias_kernel(const float* b1, const float* g1, const float* be1, const float* m1, const float* v1,
                                 const float* b2, const float* g2, const float* be2, const float* m2, const float* v2,
                                 const float* b3, const float* g3, const float* be3, const float* m3, const float* v3,
                                 float* __restrict__ biasF) {
  int i = threadIdx.x;
  if (i >= 3 * Dc) return;
  int br = i / Dc, d = i % Dc;
  const float *b, *g, *be, *mm, *v;
  if (br == 0)      { b = b1; g = g1; be = be1; mm = m1; v = v1; }
  else if (br == 1) { b = b2; g = g2; be = be2; mm = m2; v = v2; }
  else              { b = b3; g = g3; be = be3; mm = m3; v = v3; }
  float inv = g[d] * rsqrtf(v[d] + BN_EPS);
  biasF[i] = b[d] * inv + be[d] - mm[d] * inv;
}

// ---------------------------------------------------------------------------
// Async fill of one LDS slab (7 rows x 72 cols x 32 ch) from the padded
// activation tensor, via GLOBAL_LOAD_ASYNC_TO_LDS_B128 (ASYNCcnt-tracked,
// no VGPR round trip). All transfers are in-bounds thanks to halo padding.
// ---------------------------------------------------------------------------
__device__ __forceinline__ void fill_async(const __bf16* __restrict__ xP,
                                           const __bf16* lbuf,
                                           int b, int h, int cc, int tid) {
  // base of (b, h-3) row block, channel cc; unit u covers elems [8u, 8u+8)
  const __bf16* base = xP + (((long long)b * Hp + h) * Wp) * (long long)Lp + cc;
  unsigned lds_base = (unsigned)(size_t)lbuf;  // low 32 bits = LDS offset
  for (int u = tid; u < FILL_UNITS; u += 192) {
    int q  = u & 3;      // 8-channel quarter
    int rc = u >> 2;     // r*72 + col
    const __bf16* g = base + (long long)rc * Lp + q * 8;
    unsigned lds = lds_base + (unsigned)u * 16u;
    asm volatile("global_load_async_to_lds_b128 %0, %1, off"
                 :: "v"(lds), "v"(g) : "memory");
  }
}

__device__ __forceinline__ void wait_async_zero() {
  asm volatile("s_wait_asynccnt 0x0" ::: "memory");
}

// ---------------------------------------------------------------------------
// Stage 3: the WMMA conv kernel.
// One block per (b, h) output row, 192 threads = 6 waves; wave w owns
// d-channels [16w, 16w+16). Double-buffered LDS slabs; per (kh,kw) tap a
// 16x16x32 bf16 GEMM over channels, 4 pixel tiles per row.
//
// A-fragment per-lane K layout (as loaded here, 16-bit A 16x32, same K order as B):
//   lanes 0-15  (M=lane):    elems 0..15 = K 0..15 contiguous
//   lanes 16-31 (M=lane-16): elems 0..15 = K 16..31 contiguous
// B-fragment (32x16, row striped across lanes per VGPR):
//   lanes 0-15  (N=lane):    K 0..15 contiguous
//   lanes 16-31 (N=lane-16): K 16..31 contiguous
// ---------------------------------------------------------------------------
template <int S, int SHIFT, int KGBASE>
__device__ __forceinline__ void branch_mma(v8f (&acc)[4],
                                           const __bf16* __restrict__ Wt,
                                           const __bf16* tile,
                                           int d, int cc, int hi, int m) {
  for (int kh = 0; kh < S; ++kh) {
    const int row = kh + SHIFT;
    for (int kw = 0; kw < S; ++kw) {
      const int kg = KGBASE + kh * S + kw;
      // B fragment: 16 contiguous bf16 at [kg][d][cc + 16*hi]
      const __bf16* wp = Wt + (((long long)kg * Dc + d) * Lp + cc + hi * 16);
      v16bf bfrag = *(const v16bf*)wp;
#pragma unroll
      for (int wt = 0; wt < 4; ++wt) {
        const int col = wt * 16 + m + kw + SHIFT;        // halo-adjusted column
        const int base = (row * 72 + col) * 32 + hi * 16; // LDS elem index
        v8bf a0 = *(const v8bf*)(tile + base);           // K  0..7 (lo) / 16..23 (hi)
        v8bf a1 = *(const v8bf*)(tile + base + 8);       // K  8..15 (lo) / 24..31 (hi)
        v16bf afrag = __builtin_shufflevector(a0, a1,
            0, 1, 2, 3, 4, 5, 6, 7, 8, 9, 10, 11, 12, 13, 14, 15);
        acc[wt] = __builtin_amdgcn_wmma_f32_16x16x32_bf16(
            false, afrag, false, bfrag, (short)0, acc[wt], false, false);
      }
    }
  }
}

__global__ __launch_bounds__(192) void conv3_kernel(
    const __bf16* __restrict__ xP, const __bf16* __restrict__ Wt,
    const float* __restrict__ biasF,
    __bf16* __restrict__ addF, __bf16* __restrict__ mxF) {
  __shared__ __bf16 tileX[2][CHUNK_ELEMS];  // 2 x 31.5 KB double buffer

  const int b = blockIdx.x >> 6;
  const int h = blockIdx.x & 63;
  const int lane = threadIdx.x & 31;
  const int wave = threadIdx.x >> 5;  // 0..5 -> d-tile
  const int hi = lane >> 4;
  const int m = lane & 15;
  const int d = wave * 16 + m;

  v8f acc0[4], acc1[4], acc2[4];
#pragma unroll
  for (int wt = 0; wt < 4; ++wt)
#pragma unroll
    for (int r = 0; r < 8; ++r) {
      acc0[wt][r] = 0.f; acc1[wt][r] = 0.f; acc2[wt][r] = 0.f;
    }

  constexpr int NCHUNK = Lp / 32;  // 5
  // Prologue: kick off chunk 0 into buffer 0.
  fill_async(xP, tileX[0], b, h, 0, threadIdx.x);

  for (int i = 0; i < NCHUNK; ++i) {
    const int cc = i * 32;
    wait_async_zero();   // my async stores into buf (i&1) have landed
    __syncthreads();     // all waves' stores visible; buf ((i+1)&1) free
    if (i + 1 < NCHUNK)  // overlap next chunk's fill with this chunk's WMMAs
      fill_async(xP, tileX[(i + 1) & 1], b, h, cc + 32, threadIdx.x);

    const __bf16* tile = tileX[i & 1];
    branch_mma<3, 2, 0 >(acc0, Wt, tile, d, cc, hi, m);  // 3x3, pad 1
    branch_mma<5, 1, 9 >(acc1, Wt, tile, d, cc, hi, m);  // 5x5, pad 2
    branch_mma<7, 0, 34>(acc2, Wt, tile, d, cc, hi, m);  // 7x7, pad 3
  }

  // Epilogue: bias + ReLU per branch, then add / max fused maps.
  // C layout: VGPR r, lane<16 -> pixel M=r; lane>=16 -> pixel M=8+r; N=lane&15.
  const float bv1 = biasF[d], bv2 = biasF[Dc + d], bv3 = biasF[2 * Dc + d];
#pragma unroll
  for (int wt = 0; wt < 4; ++wt) {
    v8bf av, mv;
#pragma unroll
    for (int r = 0; r < 8; ++r) {
      float x1 = fmaxf(acc0[wt][r] + bv1, 0.f);
      float x2 = fmaxf(acc1[wt][r] + bv2, 0.f);
      float x3 = fmaxf(acc2[wt][r] + bv3, 0.f);
      av[r] = (__bf16)(x1 + x2 + x3);
      mv[r] = (__bf16)fmaxf(x1, fmaxf(x2, x3));
    }
    int w0 = wt * 16 + hi * 8;  // 8 consecutive pixels per lane
    long long o = (((long long)b * Dc + d) * Hs + h) * Ws + w0;
    *(v8bf*)(addF + o) = av;
    *(v8bf*)(mxF + o) = mv;
  }
}

// ---------------------------------------------------------------------------
// Stage 4: fusion conv. avg = add/3  =>  out = conv(add, wf0 + wf1/3) +
// conv(max, wf2), 3x3 pad 1, per (b,d) plane. f32 output.
// ---------------------------------------------------------------------------
__global__ void fuse_kernel(const __bf16* __restrict__ addF, const __bf16* __restrict__ mxF,
                            const float* __restrict__ wf, float* __restrict__ out) {
  long long idx = (long long)blockIdx.x * blockDim.x + threadIdx.x;
  const long long total = (long long)Bn * Dc * Hs * Ws;
  if (idx >= total) return;
  int w = (int)(idx & 63);
  int h = (int)((idx >> 6) & 63);
  long long plane = idx >> 12;
  const __bf16* ap = addF + (plane << 12);
  const __bf16* mp = mxF + (plane << 12);
  float acc = 0.f;
#pragma unroll
  for (int i = 0; i < 3; ++i) {
    int hr = h + i - 1;
    if (hr < 0 || hr >= Hs) continue;
#pragma unroll
    for (int j = 0; j < 3; ++j) {
      int wc = w + j - 1;
      if (wc < 0 || wc >= Ws) continue;
      float wa = wf[i * 3 + j] + wf[9 + i * 3 + j] * (1.f / 3.f);
      float wm = wf[18 + i * 3 + j];
      acc += (float)ap[hr * 64 + wc] * wa + (float)mp[hr * 64 + wc] * wm;
    }
  }
  out[idx] = acc;
}

// ---------------------------------------------------------------------------
extern "C" void kernel_launch(void* const* d_in, const int* in_sizes, int n_in,
                              void* d_out, int out_size, void* d_ws, size_t ws_size,
                              hipStream_t stream) {
  const float* x   = (const float*)d_in[0];
  const float* w1  = (const float*)d_in[1];
  const float* b1  = (const float*)d_in[2];
  const float* g1  = (const float*)d_in[3];
  const float* be1 = (const float*)d_in[4];
  const float* m1  = (const float*)d_in[5];
  const float* v1  = (const float*)d_in[6];
  const float* w2  = (const float*)d_in[7];
  const float* b2  = (const float*)d_in[8];
  const float* g2  = (const float*)d_in[9];
  const float* be2 = (const float*)d_in[10];
  const float* m2  = (const float*)d_in[11];
  const float* v2  = (const float*)d_in[12];
  const float* w3  = (const float*)d_in[13];
  const float* b3  = (const float*)d_in[14];
  const float* g3  = (const float*)d_in[15];
  const float* be3 = (const float*)d_in[16];
  const float* m3  = (const float*)d_in[17];
  const float* v3  = (const float*)d_in[18];
  const float* wf  = (const float*)d_in[19];

  char* ws = (char*)d_ws;
  size_t off = 0;
  auto alloc = [&](size_t bytes) {
    void* p = ws + off;
    off = (off + bytes + 255) & ~(size_t)255;
    return p;
  };
  __bf16* xP    = (__bf16*)alloc((size_t)Bn * Hp * Wp * Lp * 2);       // 51.6 MB
  __bf16* Wt    = (__bf16*)alloc((size_t)KG_TOTAL * Dc * Lp * 2);      //  2.6 MB
  float*  biasF = (float*) alloc((size_t)3 * Dc * 4);
  __bf16* addF  = (__bf16*)alloc((size_t)Bn * Dc * Hs * Ws * 2);       // 25.2 MB
  __bf16* mxF   = (__bf16*)alloc((size_t)Bn * Dc * Hs * Ws * 2);       // 25.2 MB

  {
    long long total = (long long)Bn * Hp * Wp * Lp;
    prep_x_kernel<<<(unsigned)((total + 255) / 256), 256, 0, stream>>>(x, xP);
  }
  {
    int total = KG_TOTAL * Dc * Lp;
    prep_w_kernel<<<(total + 255) / 256, 256, 0, stream>>>(
        w1, g1, v1, w2, g2, v2, w3, g3, v3, Wt);
  }
  prep_bias_kernel<<<1, 288, 0, stream>>>(
      b1, g1, be1, m1, v1, b2, g2, be2, m2, v2, b3, g3, be3, m3, v3, biasF);

  conv3_kernel<<<Bn * Hs, 192, 0, stream>>>(xP, Wt, biasF, addF, mxF);

  {
    long long total = (long long)Bn * Dc * Hs * Ws;
    fuse_kernel<<<(unsigned)((total + 255) / 256), 256, 0, stream>>>(
        addF, mxF, wf, (float*)d_out);
  }
}
